// DAPPM_23922967838996
// MI455X (gfx1250) — matrix operand order repaired and checked by reading, not compiled, not run on hardware
//
#include <hip/hip_runtime.h>

// ---------------------------------------------------------------------------
// DAPPM sparse block for MI455X (gfx1250, wave32, WMMA).
// Roofline: conv gathers are L2-bound -> store activations bf16 (halves bytes),
// all matmuls on v_wmma_f32_16x16x32_bf16 with fp32 accumulation.
// ---------------------------------------------------------------------------

#define N_PTS   200000
#define C_IN    128
#define CBR     64
#define C_OUT   128
#define KTAPS   27
#define LD_CAT  320          // 5 * CBR
#define EPSBN   1e-5f
#define STATS_BLOCKS 512

typedef __attribute__((ext_vector_type(16))) __bf16 v16bf;
typedef __attribute__((ext_vector_type(8)))  __bf16 v8bf;
typedef __attribute__((ext_vector_type(8)))  float  v8f;

static __device__ __forceinline__ __bf16 f2bf(float f) {
    union { float f; unsigned u; } cv; cv.f = f;
    unsigned u = cv.u;
    u = u + 0x7FFFu + ((u >> 16) & 1u);          // round-to-nearest-even
    unsigned short h = (unsigned short)(u >> 16);
    union { unsigned short s; __bf16 b; } o; o.s = h;
    return o.b;
}
static __device__ __forceinline__ float bf2f(__bf16 b) {
    union { __bf16 b; unsigned short s; } i; i.b = b;
    union { unsigned u; float f; } o; o.u = ((unsigned)i.s) << 16;
    return o.f;
}
static __device__ __forceinline__ v16bf concat8(v8bf lo, v8bf hi) {
    v16bf r;
#pragma unroll
    for (int e = 0; e < 8; ++e) { r[e] = lo[e]; r[e + 8] = hi[e]; }
    return r;
}

// ---------------------------------------------------------------------------
// Pack fp32 row-major W[Cin][Cout] into WMMA B-fragment order (bf16).
// B 32x16 bf16 layout: lane%16 = N, elements i = K (table order, h=lane/16).
// pack[((kc*NT + nt)*32 + lane)*16 + i] = W[kc*32 + (i<8 ? 8h+i : 16+8h+i-8)][nt*16 + lane%16]
// ---------------------------------------------------------------------------
__global__ void dappm_pack_b(const float* __restrict__ W, __bf16* __restrict__ pack,
                             int Cin, int Cout, int nMat) {
    long gid = (long)blockIdx.x * blockDim.x + threadIdx.x;
    long per = (long)Cin * Cout;
    if (gid >= per * nMat) return;
    int m = (int)(gid / per);
    int e = (int)(gid % per);
    int NT = Cout >> 4;
    int i    = e & 15;
    int lane = (e >> 4) & 31;
    int nt   = (e >> 9) % NT;
    int kc   = e / (512 * NT);
    int k = kc * 32 + (i < 8 ? (lane >> 4) * 8 + i : 16 + (lane >> 4) * 8 + (i - 8));
    int n = nt * 16 + (lane & 15);
    pack[(long)m * per + e] = f2bf(W[(long)m * per + (long)k * Cout + n]);
}

// ---------------------------------------------------------------------------
// Deterministic two-stage column statistics (training-mode BatchNorm).
// ---------------------------------------------------------------------------
__global__ void dappm_colstats_partial(const void* __restrict__ X, int isBf16,
                                       int rows, int C, float* __restrict__ partial) {
    int blk = blockIdx.x;
    int rowsPer = (rows + STATS_BLOCKS - 1) / STATS_BLOCKS;
    int r0 = blk * rowsPer;
    int r1 = r0 + rowsPer; if (r1 > rows) r1 = rows;
    for (int c = threadIdx.x; c < C; c += blockDim.x) {
        float s = 0.f, q = 0.f;
        if (isBf16) {
            const __bf16* xb = (const __bf16*)X;
            for (int r = r0; r < r1; ++r) { float v = bf2f(xb[(long)r * C + c]); s += v; q += v * v; }
        } else {
            const float* xf = (const float*)X;
            for (int r = r0; r < r1; ++r) { float v = xf[(long)r * C + c]; s += v; q += v * v; }
        }
        partial[(long)blk * 2 * C + c]     = s;
        partial[(long)blk * 2 * C + C + c] = q;
    }
}

__global__ void dappm_colstats_finalize(const float* __restrict__ partial, int rows, int C,
                                        const float* __restrict__ gamma,
                                        const float* __restrict__ beta,
                                        float* __restrict__ ab) {
    int c = blockIdx.x * blockDim.x + threadIdx.x;
    if (c >= C) return;
    float s = 0.f, q = 0.f;
    for (int b = 0; b < STATS_BLOCKS; ++b) {
        s += partial[(long)b * 2 * C + c];
        q += partial[(long)b * 2 * C + C + c];
    }
    float inv = 1.0f / (float)rows;
    float mu  = s * inv;
    float var = q * inv - mu * mu; if (var < 0.f) var = 0.f;
    float a = gamma[c] * rsqrtf(var + EPSBN);
    ab[c]     = a;
    ab[C + c] = beta[c] - mu * a;
}

// y = bf16(relu(x*a + b)); input fp32 or bf16; safe in place.
__global__ void dappm_bn_relu_cvt(const void* __restrict__ X, int isBf16,
                                  const float* __restrict__ ab, int C, long total,
                                  __bf16* __restrict__ out) {
    long gid = (long)blockIdx.x * blockDim.x + threadIdx.x;
    if (gid >= total) return;
    int c = (int)(gid % C);
    float v = isBf16 ? bf2f(((const __bf16*)X)[gid]) : ((const float*)X)[gid];
    float r = v * ab[c] + ab[C + c];
    out[gid] = f2bf(r > 0.f ? r : 0.f);
}

// ---------------------------------------------------------------------------
// Sparse average pooling (segment mean) via fp32 atomics.
// ---------------------------------------------------------------------------
__global__ void dappm_segpool_add(const float* __restrict__ feats, const int* __restrict__ idx,
                                  float* __restrict__ psum, float* __restrict__ cnt,
                                  int rows, int C) {
    long gid = (long)blockIdx.x * blockDim.x + threadIdx.x;
    if (gid >= (long)rows * C) return;
    int r = (int)(gid / C), c = (int)(gid % C);
    int seg = idx[r];
    atomicAdd(&psum[(long)seg * C + c], feats[gid]);
    if (c == 0) atomicAdd(&cnt[seg], 1.0f);
}

__global__ void dappm_segfinish(float* __restrict__ psum, const float* __restrict__ cnt,
                                int M, int C) {
    long gid = (long)blockIdx.x * blockDim.x + threadIdx.x;
    if (gid >= (long)M * C) return;
    float cc = cnt[gid / C];
    psum[gid] = psum[gid] / fmaxf(cc, 1.0f);
}

// z = t[up[r]] + xs_prev[r]  (gather of pooled 1x1-conv output + running branch)
__global__ void dappm_zadd(const float* __restrict__ t, const int* __restrict__ up,
                           const __bf16* __restrict__ xs_prev, int ldxs,
                           float* __restrict__ z, int rows) {
    long gid = (long)blockIdx.x * blockDim.x + threadIdx.x;
    if (gid >= (long)rows * CBR) return;
    int r = (int)(gid / CBR), c = (int)(gid % CBR);
    z[gid] = t[(long)up[r] * CBR + c] + bf2f(xs_prev[(long)r * ldxs + c]);
}

// ---------------------------------------------------------------------------
// Generic WMMA GEMM: Y[rows,Cout] (=|+=) U[rows,Cin]bf16 @ packB.
// One wave per (16-row, 16-col) tile; 8 waves per block; fp32 accumulators.
// A frag (16x32 bf16): lane holds row M=lane%16; K runs {kc*32+8h..+7} and
// {kc*32+16+8h..+7} with h=lane/16 -> two 16B contiguous loads.
// ---------------------------------------------------------------------------
__global__ __launch_bounds__(256)
void dappm_gemm_wmma(const __bf16* __restrict__ U, const __bf16* __restrict__ pack,
                     int rows, int Cin, int Cout,
                     float* __restrict__ Yf, __bf16* __restrict__ Yb,
                     int ldY, int accumulate) {
    int lane = threadIdx.x & 31;
    int wave = threadIdx.x >> 5;
    int rowTile = blockIdx.x * 8 + wave;
    int numRowTiles = (rows + 15) >> 4;
    if (rowTile >= numRowTiles) return;          // wave-uniform
    int nt = blockIdx.y;
    int NT = Cout >> 4;
    int lo = lane & 15, hi = lane >> 4;
    int arow = rowTile * 16 + lo;
    if (arow >= rows) arow = rows - 1;           // clamp tail reads
    const __bf16* urow = U + (long)arow * Cin;

    v8f acc = {0.f,0.f,0.f,0.f,0.f,0.f,0.f,0.f};
    int KC = Cin >> 5;
    for (int kc = 0; kc < KC; ++kc) {
        v8bf a0 = *(const v8bf*)(urow + kc * 32 + hi * 8);
        v8bf a1 = *(const v8bf*)(urow + kc * 32 + 16 + hi * 8);
        v16bf a = concat8(a0, a1);
        const __bf16* bp = pack + ((long)(kc * NT + nt) * 32 + lane) * 16;
        v16bf b = concat8(*(const v8bf*)bp, *(const v8bf*)(bp + 8));
        acc = __builtin_amdgcn_wmma_f32_16x16x32_bf16(false, a, false, b,
                                                      (short)0, acc, false, false);
    }
#pragma unroll
    for (int r = 0; r < 8; ++r) {
        int m = rowTile * 16 + hi * 8 + r;       // C/D layout: VGPR r -> M=r+8*hi
        if (m < rows) {
            long o = (long)m * ldY + nt * 16 + lo;
            if (Yb)            Yb[o] = f2bf(acc[r]);
            else if (accumulate) Yf[o] += acc[r];
            else               Yf[o] = acc[r];
        }
    }
}

// ---------------------------------------------------------------------------
// 27-tap sparse conv: Y[r] = sum_j gather(U, nbr[r,j]) @ Wj  (64->64, bf16 WMMA)
// Weights double-buffered tap-by-tap through LDS (8 KB/tap), shared by 8 waves.
// ---------------------------------------------------------------------------
__global__ __launch_bounds__(256)
void dappm_conv27_wmma(const __bf16* __restrict__ U, const int* __restrict__ nbr,
                       const __bf16* __restrict__ pack, __bf16* __restrict__ Y,
                       int ldY, int rows) {
    __shared__ __bf16 smem[2][4096];             // 2 x 8KB tap weights
    int tid  = threadIdx.x;
    int lane = tid & 31;
    int wave = tid >> 5;
    int rowTile = blockIdx.x * 8 + wave;
    int active = (rowTile * 16 < rows);          // wave-uniform
    int lo = lane & 15, hi = lane >> 4;
    int myrow = rowTile * 16 + lo;

    v8f acc[4];
#pragma unroll
    for (int nt = 0; nt < 4; ++nt) acc[nt] = (v8f){0.f,0.f,0.f,0.f,0.f,0.f,0.f,0.f};

    { // stage tap 0
        const v8bf* g = (const v8bf*)pack;
        v8bf* s = (v8bf*)smem[0];
        s[tid] = g[tid]; s[tid + 256] = g[tid + 256];
    }
    __syncthreads();

    for (int j = 0; j < KTAPS; ++j) {
        int cur = j & 1;
        if (j + 1 < KTAPS) {                     // prefetch next tap into other buffer
            const v8bf* g = (const v8bf*)(pack + (long)(j + 1) * 4096);
            v8bf* s = (v8bf*)smem[cur ^ 1];
            s[tid] = g[tid]; s[tid + 256] = g[tid + 256];
        }
        if (active) {
            int idx = nbr[myrow * KTAPS + j];
            if (j + 1 < KTAPS) __builtin_prefetch(&nbr[myrow * KTAPS + j + 1], 0, 0);
            const __bf16* urow = U + (long)idx * CBR;
#pragma unroll
            for (int kc = 0; kc < 2; ++kc) {
                v8bf a0 = *(const v8bf*)(urow + kc * 32 + hi * 8);
                v8bf a1 = *(const v8bf*)(urow + kc * 32 + 16 + hi * 8);
                v16bf a = concat8(a0, a1);
#pragma unroll
                for (int nt = 0; nt < 4; ++nt) {
                    const v8bf* bp = (const v8bf*)smem[cur] + ((kc * 4 + nt) * 32 + lane) * 2;
                    v16bf b = concat8(bp[0], bp[1]);
                    acc[nt] = __builtin_amdgcn_wmma_f32_16x16x32_bf16(
                        false, a, false, b, (short)0, acc[nt], false, false);
                }
            }
        }
        __syncthreads();
    }

    if (active) {
#pragma unroll
        for (int nt = 0; nt < 4; ++nt)
#pragma unroll
            for (int r = 0; r < 8; ++r) {
                int m = rowTile * 16 + hi * 8 + r;
                Y[(long)m * ldY + nt * 16 + lo] = f2bf(acc[nt][r]);
            }
    }
}

// ---------------------------------------------------------------------------
static inline int ceilDivL(long a, long b) { return (int)((a + b - 1) / b); }

extern "C" void kernel_launch(void* const* d_in, const int* in_sizes, int n_in,
                              void* d_out, int out_size, void* d_ws, size_t ws_size,
                              hipStream_t stream) {
    (void)in_sizes; (void)n_in; (void)out_size; (void)ws_size;
    const float* feats         = (const float*)d_in[0];
    const float* bn_in_gamma   = (const float*)d_in[1];
    const float* bn_in_beta    = (const float*)d_in[2];
    const float* w_scale       = (const float*)d_in[3];
    const float* bn_proc_gamma = (const float*)d_in[4];
    const float* bn_proc_beta  = (const float*)d_in[5];
    const float* w_proc        = (const float*)d_in[6];
    const float* bn_comp_gamma = (const float*)d_in[7];
    const float* bn_comp_beta  = (const float*)d_in[8];
    const float* w_comp        = (const float*)d_in[9];
    const float* bn_sc_gamma   = (const float*)d_in[10];
    const float* bn_sc_beta    = (const float*)d_in[11];
    const float* w_sc          = (const float*)d_in[12];
    const int*   pool_idx[4]   = {(const int*)d_in[13], (const int*)d_in[15],
                                  (const int*)d_in[17], (const int*)d_in[19]};
    const int*   up_idx[4]     = {(const int*)d_in[14], (const int*)d_in[16],
                                  (const int*)d_in[18], (const int*)d_in[20]};
    const int*   nbr           = (const int*)d_in[21];
    float*       out           = (float*)d_out;
    static const int MSC[4] = {50000, 12500, 3125, 800};

    // Workspace bump allocator (256B aligned).
    char* base = (char*)d_ws; size_t off = 0;
    auto alloc = [&](size_t bytes) -> void* {
        void* p = base + off;
        off = (off + bytes + 255) & ~(size_t)255;
        return p;
    };
    float*  ab_in0  = (float*)alloc(2 * C_IN * 4);
    float*  ab_sc   = (float*)alloc(2 * C_IN * 4);
    float*  ab_cur  = (float*)alloc(2 * LD_CAT * 4);
    float*  partial = (float*)alloc((size_t)STATS_BLOCKS * 2 * LD_CAT * 4);
    __bf16* packScale = (__bf16*)alloc((size_t)5 * C_IN * CBR * 2);
    __bf16* packConv  = (__bf16*)alloc((size_t)4 * KTAPS * CBR * CBR * 2);
    __bf16* packComp  = (__bf16*)alloc((size_t)LD_CAT * C_OUT * 2);
    __bf16* packSc    = (__bf16*)alloc((size_t)C_IN * C_OUT * 2);
    __bf16* Ufeat = (__bf16*)alloc((size_t)N_PTS * C_IN * 2);
    __bf16* xs    = (__bf16*)alloc((size_t)N_PTS * LD_CAT * 2);
    float*  zbuf  = (float*)alloc((size_t)N_PTS * CBR * 4);
    __bf16* ubuf  = (__bf16*)alloc((size_t)N_PTS * CBR * 2);
    float*  psum  = (float*)alloc((size_t)50000 * C_IN * 4);
    float*  cnt   = (float*)alloc((size_t)50000 * 4);
    __bf16* p_bf  = (__bf16*)alloc((size_t)50000 * C_IN * 2);
    float*  tbuf  = (float*)alloc((size_t)50000 * CBR * 4);

    // --- Pack all weights into WMMA B-fragment order (bf16) ---
    dappm_pack_b<<<ceilDivL(5L * C_IN * CBR, 256), 256, 0, stream>>>(w_scale, packScale, C_IN, CBR, 5);
    dappm_pack_b<<<ceilDivL(108L * CBR * CBR, 256), 256, 0, stream>>>(w_proc, packConv, CBR, CBR, 108);
    dappm_pack_b<<<ceilDivL((long)LD_CAT * C_OUT, 256), 256, 0, stream>>>(w_comp, packComp, LD_CAT, C_OUT, 1);
    dappm_pack_b<<<ceilDivL((long)C_IN * C_OUT, 256), 256, 0, stream>>>(w_sc, packSc, C_IN, C_OUT, 1);

    // --- BN stats of feats (shared by bn_in[0] and bn_sc) ---
    dappm_colstats_partial<<<STATS_BLOCKS, 256, 0, stream>>>(feats, 0, N_PTS, C_IN, partial);
    dappm_colstats_finalize<<<1, C_IN, 0, stream>>>(partial, N_PTS, C_IN, bn_in_gamma, bn_in_beta, ab_in0);
    dappm_colstats_finalize<<<1, C_IN, 0, stream>>>(partial, N_PTS, C_IN, bn_sc_gamma, bn_sc_beta, ab_sc);

    // --- scale0: bn_relu(feats) @ w_scale[0] -> xs[:, 0:64] (bf16) ---
    long tf = (long)N_PTS * C_IN;
    dappm_bn_relu_cvt<<<ceilDivL(tf, 256), 256, 0, stream>>>(feats, 0, ab_in0, C_IN, tf, Ufeat);
    {
        dim3 g(ceilDivL((N_PTS + 15) / 16, 8), CBR / 16);
        dappm_gemm_wmma<<<g, 256, 0, stream>>>(Ufeat, packScale, N_PTS, C_IN, CBR,
                                               nullptr, xs, LD_CAT, 0);
    }

    // --- 4 scales: pool -> bn_relu -> 1x1 -> gather+add -> bn_relu -> 27-tap conv ---
    for (int k = 0; k < 4; ++k) {
        int M = MSC[k];
        hipMemsetAsync(psum, 0, (size_t)M * C_IN * 4, stream);
        hipMemsetAsync(cnt, 0, (size_t)M * 4, stream);
        dappm_segpool_add<<<ceilDivL(tf, 256), 256, 0, stream>>>(feats, pool_idx[k], psum, cnt, N_PTS, C_IN);
        long tm = (long)M * C_IN;
        dappm_segfinish<<<ceilDivL(tm, 256), 256, 0, stream>>>(psum, cnt, M, C_IN);

        dappm_colstats_partial<<<STATS_BLOCKS, 256, 0, stream>>>(psum, 0, M, C_IN, partial);
        dappm_colstats_finalize<<<1, C_IN, 0, stream>>>(partial, M, C_IN,
                                                        bn_in_gamma + (k + 1) * C_IN,
                                                        bn_in_beta + (k + 1) * C_IN, ab_cur);
        dappm_bn_relu_cvt<<<ceilDivL(tm, 256), 256, 0, stream>>>(psum, 0, ab_cur, C_IN, tm, p_bf);
        {
            dim3 g(ceilDivL((M + 15) / 16, 8), CBR / 16);
            dappm_gemm_wmma<<<g, 256, 0, stream>>>(p_bf, packScale + (size_t)(k + 1) * C_IN * CBR,
                                                   M, C_IN, CBR, tbuf, nullptr, CBR, 0);
        }
        long tz = (long)N_PTS * CBR;
        dappm_zadd<<<ceilDivL(tz, 256), 256, 0, stream>>>(tbuf, up_idx[k], xs + (size_t)k * CBR,
                                                          LD_CAT, zbuf, N_PTS);
        dappm_colstats_partial<<<STATS_BLOCKS, 256, 0, stream>>>(zbuf, 0, N_PTS, CBR, partial);
        dappm_colstats_finalize<<<1, CBR, 0, stream>>>(partial, N_PTS, CBR,
                                                       bn_proc_gamma + k * CBR,
                                                       bn_proc_beta + k * CBR, ab_cur);
        dappm_bn_relu_cvt<<<ceilDivL(tz, 256), 256, 0, stream>>>(zbuf, 0, ab_cur, CBR, tz, ubuf);
        dappm_conv27_wmma<<<ceilDivL(N_PTS / 16, 8), 256, 0, stream>>>(
            ubuf, nbr, packConv + (size_t)k * KTAPS * CBR * CBR,
            xs + (size_t)(k + 1) * CBR, LD_CAT, N_PTS);
    }

    // --- compose: bn_relu(cat) @ w_comp + bn_relu(feats) @ w_sc ---
    dappm_colstats_partial<<<STATS_BLOCKS, 256, 0, stream>>>(xs, 1, N_PTS, LD_CAT, partial);
    dappm_colstats_finalize<<<1, LD_CAT, 0, stream>>>(partial, N_PTS, LD_CAT,
                                                      bn_comp_gamma, bn_comp_beta, ab_cur);
    long tc = (long)N_PTS * LD_CAT;
    dappm_bn_relu_cvt<<<ceilDivL(tc, 256), 256, 0, stream>>>(xs, 1, ab_cur, LD_CAT, tc, xs); // in place
    {
        dim3 g(ceilDivL((N_PTS + 15) / 16, 8), C_OUT / 16);
        dappm_gemm_wmma<<<g, 256, 0, stream>>>(xs, packComp, N_PTS, LD_CAT, C_OUT,
                                               out, nullptr, C_OUT, 0);
        dappm_bn_relu_cvt<<<ceilDivL(tf, 256), 256, 0, stream>>>(feats, 0, ab_sc, C_IN, tf, Ufeat);
        dappm_gemm_wmma<<<g, 256, 0, stream>>>(Ufeat, packSc, N_PTS, C_IN, C_OUT,
                                               out, nullptr, C_OUT, 1);
    }
}
